// RNN_50534585205210
// MI455X (gfx1250) — matrix operand order repaired and checked by reading, not compiled, not run on hardware
//
#include <hip/hip_runtime.h>
#include <hip/hip_bf16.h>
#include <stdint.h>

// ---------------- problem dimensions ----------------
#define HID   1632      // 3*16*17*2
#define H3    4896      // 3*HID
#define NST   16
#define NRES  4
#define NRAW  104
#define NMEAS 8
#define TT    2000
#define NTRI_ 136
#define OUTD  3264      // OUT = HID/2 * NRES
#define LN_N  6528      // 2*OUT
#define NT    1024      // scan block threads (32 wave32s -> 1 WGP)
#define KPAD  128       // GEMM K padding (104 -> 128)

typedef __attribute__((ext_vector_type(16))) __bf16 v16bf;
typedef __attribute__((ext_vector_type(8)))  __bf16 v8bf;
typedef __attribute__((ext_vector_type(8)))  float  v8f;
typedef __attribute__((ext_vector_type(4)))  unsigned int uint4v;
typedef __attribute__((ext_vector_type(8)))  int int8v;
typedef __attribute__((ext_vector_type(4)))  int int4v;

#if defined(__has_builtin)
#if __has_builtin(__builtin_amdgcn_tensor_load_to_lds) && __has_builtin(__builtin_amdgcn_s_wait_tensorcnt)
#define HAVE_TDM 1
#endif
#endif
#ifndef HAVE_TDM
#define HAVE_TDM 0
#endif

__device__ __forceinline__ unsigned short f2bf_bits(float f) {
  unsigned u = __float_as_uint(f);
  return (unsigned short)((u + 0x7fffu + ((u >> 16) & 1u)) >> 16);  // RNE
}
__device__ __forceinline__ float bf_bits2f(unsigned short s) {
  return __uint_as_float(((unsigned)s) << 16);
}

// ---------------------------------------------------------------------------
// One-time: fp32 (O x K, row major) -> bf16 bits transposed (K x O).
// Transposed so the scan mat-vecs read consecutive output rows per lane
// (coalesced 64B/wave loads; all bf16 weights total ~93MB -> resident in
// the 192MB L2, so the 2000-step recurrence streams from L2 not HBM).
// ---------------------------------------------------------------------------
__global__ void conv_tr_kernel(const float* __restrict__ src,
                               unsigned short* __restrict__ dst,
                               int O, int K) {
  size_t idx = (size_t)blockIdx.x * blockDim.x + threadIdx.x;
  size_t n = (size_t)O * K;
  if (idx >= n) return;
  int o = (int)(idx / K), k = (int)(idx % K);
  dst[(size_t)k * O + o] = f2bf_bits(src[(size_t)o * K + k]);
}

// One-time: fp32 (R x srcK row major) -> bf16 row major (R x dstK), zero pad.
// Produces layout-matched, K-padded GEMM operands so WMMA fragments are
// assembled from contiguous b128 loads with no masking / conversion VALU.
__global__ void stage_pad_kernel(const float* __restrict__ src,
                                 unsigned short* __restrict__ dst,
                                 int R, int srcK, int dstK, int useK) {
  size_t idx = (size_t)blockIdx.x * blockDim.x + threadIdx.x;
  size_t n = (size_t)R * dstK;
  if (idx >= n) return;
  int r = (int)(idx / dstK), k = (int)(idx % dstK);
  dst[idx] = (k < useK) ? f2bf_bits(src[(size_t)r * srcK + k]) : (unsigned short)0;
}

// ---------------------------------------------------------------------------
// theta_feat = lin1_w @ relu(lin0_w @ theta + lin0_b) + lin1_b  (one-time)
// ---------------------------------------------------------------------------
__global__ void theta_feat_kernel(const float* __restrict__ theta,
                                  const float* __restrict__ l0w, const float* __restrict__ l0b,
                                  const float* __restrict__ l1w, const float* __restrict__ l1b,
                                  float* __restrict__ tf) {
  __shared__ float th[16];
  __shared__ float hid[HID];
  int tid = threadIdx.x;
  if (tid < 16) th[tid] = theta[tid];
  __syncthreads();
  for (int h = tid; h < HID; h += blockDim.x) {
    float acc = l0b[h];
    for (int k = 0; k < 16; ++k) acc += l0w[h * 16 + k] * th[k];
    hid[h] = fmaxf(acc, 0.f);
  }
  __syncthreads();
  for (int o = tid; o < OUTD; o += blockDim.x) {
    float acc = l1b[o];
    for (int k = 0; k < HID; ++k) acc += l1w[(size_t)o * HID + k] * hid[k];
    tf[o] = acc;
  }
}

// ---------------------------------------------------------------------------
// WMMA GEMM: ig0pre[t][r] = sum_{k<104} w0ih[r][k] * data_seq[t][k]
// M=4896 (306 tiles), N=2000 (125 tiles), K staged/padded to 128 (4x k32).
// One wave per 16x16 tile; v_wmma_f32_16x16x32_bf16, f32 accumulate.
// Fragment k-index math per CDNA5 ISA 7.12.2: per lane the A fragment needs
// k = k0+8*hi+{0..7} and k0+16+8*hi+{0..7}; B uses the same k as A --
// all contiguous in the staged row-major padded buffers -> b128 loads.
// ---------------------------------------------------------------------------
#define GEMM_MT 306
#define GEMM_NT 125

__global__ void ig0_gemm_kernel(const unsigned short* __restrict__ aG,   // [r][128] bf16
                                const unsigned short* __restrict__ bG,   // [t][128] bf16
                                float* __restrict__ ig0pre) {            // [t][4896]
  int wave = blockIdx.x * (blockDim.x >> 5) + (threadIdx.x >> 5);
  if (wave >= GEMM_MT * GEMM_NT) return;        // wave-uniform: EXEC stays all-ones
  int tm = wave % GEMM_MT;
  int tn = wave / GEMM_MT;
  int lane = threadIdx.x & 31;
  int hi = lane >> 4, lm = lane & 15;

  const __bf16* arow = (const __bf16*)(aG + (size_t)(tm * 16 + lm) * KPAD);
  const __bf16* brow = (const __bf16*)(bG + (size_t)(tn * 16 + lm) * KPAD);

  v8f c = {};
#pragma unroll
  for (int kb = 0; kb < 4; ++kb) {
    int k0 = kb * 32;
    v8bf alo = *(const v8bf*)(arow + k0 + hi * 8);        // A: v=0..3 (8 consec k)
    v8bf ahi = *(const v8bf*)(arow + k0 + 16 + hi * 8);   // A: v=4..7
    v8bf blo = *(const v8bf*)(brow + k0 + hi * 8);        // B: v=0..3 (matches A)
    v8bf bhi = *(const v8bf*)(brow + k0 + 16 + hi * 8);   // B: v=4..7
    v16bf a, b;
#pragma unroll
    for (int i = 0; i < 8; ++i) {
      a[i] = alo[i]; a[8 + i] = ahi[i];
      b[i] = blo[i]; b[8 + i] = bhi[i];
    }
    c = __builtin_amdgcn_wmma_f32_16x16x32_bf16(false, a, false, b,
                                                (short)0, c, false, false);
  }
#pragma unroll
  for (int r = 0; r < 8; ++r) {
    int rowm = tm * 16 + r + 8 * hi;            // D: m = r + 8*hi, n = lane%16
    int tcol = tn * 16 + lm;
    ig0pre[(size_t)tcol * H3 + rowm] = c[r];
  }
}

// ---------------------------------------------------------------------------
// TDM: async-load one ig0 column (H3 f32, one row) from global to LDS.
// D# per CDNA5 ISA 8.3/8.4: group0 = {count=1, lds_addr, global_addr, type=2};
// group1 = {data_size=4B, tensor_dim0=tile_dim0=H3, tensor_dim1=tile_dim1=1}.
// Groups 2/3 unused (<=2D tensor); this toolchain's builtin takes the 6-arg
// form with a trailing int32x8 group + cache-policy word.
// Issued by wave 0 only; completion gated by s_wait_tensorcnt + barrier.
// ---------------------------------------------------------------------------
__device__ __forceinline__ void tdm_load_col(const float* gaddr, unsigned lds_off) {
#if HAVE_TDM
  unsigned long long ga = (unsigned long long)(uintptr_t)gaddr;
  uint4v g0;
  g0[0] = 1u;                                           // count=1, user mode
  g0[1] = lds_off;                                      // lds_addr (bytes)
  g0[2] = (unsigned)(ga & 0xFFFFFFFFu);                 // global_addr[31:0]
  g0[3] = (unsigned)((ga >> 32) & 0x1FFFFFFu)           // global_addr[56:32]
        | (2u << 30);                                   // type=2 ("image")
  int8v g1;
  g1[0] = (int)(2u << 16);                              // data_size=2 (4 bytes)
  g1[1] = (int)(((unsigned)H3 & 0xFFFFu) << 16);        // tensor_dim0[15:0] @bits63:48
  g1[2] = (int)((((unsigned)H3 >> 16) & 0xFFFFu)        // tensor_dim0[31:16]
        | (1u << 16));                                  // tensor_dim1=1 (low half)
  g1[3] = (int)(((unsigned)H3 & 0xFFFFu) << 16);        // tile_dim0 @bits127:112
  g1[4] = 1;                                            // tile_dim1=1
  g1[5] = H3;                                           // tensor_dim0_stride
  g1[6] = 0; g1[7] = 0;
  int4v z4 = {0, 0, 0, 0};
  int8v z8 = {0, 0, 0, 0, 0, 0, 0, 0};
  __builtin_amdgcn_tensor_load_to_lds(g0, g1, z4, z4, z8, 0);
#else
  (void)gaddr; (void)lds_off;
#endif
}

// ---------------------------------------------------------------------------
// Persistent sequential scan: one 1024-thread workgroup, 2000 steps.
// bf16 weights streamed from L2, fp32 accumulation, Kalman filter in fp32.
// ig0 columns double-buffered in LDS via TDM (fallback: prefetch + global).
// ---------------------------------------------------------------------------
__global__ void __launch_bounds__(NT, 1)
rnn_scan_kernel(const float* __restrict__ ig0pre,
                const unsigned short* __restrict__ w0ihT,
                const unsigned short* __restrict__ w0hhT,
                const unsigned short* __restrict__ w1ihT,
                const unsigned short* __restrict__ w1hhT,
                const unsigned short* __restrict__ linwT,
                const unsigned short* __restrict__ f1wT,
                const unsigned short* __restrict__ f2wT,
                const float* __restrict__ tf,
                const float* __restrict__ b0,  const float* __restrict__ bn0,
                const float* __restrict__ b1,  const float* __restrict__ bn1,
                const float* __restrict__ lin_b,
                const float* __restrict__ flnw, const float* __restrict__ flnb,
                const float* __restrict__ f1b,  const float* __restrict__ f2b,
                const float* __restrict__ Hm,   const float* __restrict__ Rm,
                const float* __restrict__ ym,
                const float* __restrict__ msi,  const float* __restrict__ vsi,
                float* __restrict__ outbuf) {
  const int tid = threadIdx.x;
  extern __shared__ float sm[];
  float* h0   = sm;             // 1632
  float* h1   = h0 + HID;       // 1632
  float* bufA = h1 + HID;       // 4896
  float* bufB = bufA + H3;      // 4896
  float* outv = bufB + H3;      // 6528
  float* hidv = outv + LN_N;    // 1632
  float* out2 = hidv + HID;     // 1632
  float* inps = out2 + HID;     // 152 (pad 160)
  float* red  = inps + 160;     // 1024
  float* km_  = red + NT;       // 16
  float* kv_  = km_ + 16;       // 256
  float* mp_  = kv_ + 256;      // 16
  float* vp_  = mp_ + 16;       // 256
  float* t1_  = vp_ + 256;      // 256
  float* Lr_  = t1_ + 256;      // 256
  float* vs_  = Lr_ + 256;      // 256
  float* S_   = vs_ + 256;      // 64
  float* C_   = S_ + 64;        // 128
  float* Kg_  = C_ + 128;       // 128
  float* rs_  = Kg_ + 128;      // 8
  float* mn_  = rs_ + 8;        // 16
  float* vn_  = mn_ + 16;       // 256
  float* igbuf = vn_ + 256;     // 2*4896 (TDM double buffer)

  // output sections (flat concat order of the reference tuple)
  float* fm = outbuf;                 // filt_mean (8001,16)
  float* fv = outbuf + 128016;        // filt_var  (8001,16,16)
  float* pm = outbuf + 2176272;       // pred_mean (8000,16)
  float* pv = outbuf + 2304272;       // pred_var  (8000,16,16)
  float* wg = outbuf + 4352272;       // wgt       (8000,16,16)

  for (int j = tid; j < HID; j += NT) { h0[j] = 0.f; h1[j] = 0.f; }
  if (tid < 16)  { km_[tid] = msi[tid]; fm[tid] = msi[tid]; }
  if (tid < 256) { kv_[tid] = vsi[tid]; fv[tid] = vsi[tid]; }

#if HAVE_TDM
  const unsigned ig_lds0 = (unsigned)((uintptr_t)((char*)igbuf - (char*)sm));
  if ((tid >> 5) == 0) tdm_load_col(ig0pre, ig_lds0);   // preload t=0
#endif
  __syncthreads();

  for (int t = 0; t < TT; ++t) {
#if HAVE_TDM
    if ((tid >> 5) == 0) __builtin_amdgcn_s_wait_tensorcnt(0); // buf[t&1] ready
#endif
    // state-dependent part of the GRU0 input: [m_filt, triu(v_filt)]
    if (tid < 152) {
      float val;
      if (tid < 16) val = km_[tid];
      else {
        int u = tid - 16, i = 0;
        while (u >= 16 - i) { u -= 16 - i; ++i; }
        val = kv_[i * 16 + (i + u)];
      }
      inps[tid] = val;
    }
#if !HAVE_TDM
    if (t + 1 < TT) {   // fallback: prefetch next column (global_prefetch_b8)
      const float* nx = ig0pre + (size_t)(t + 1) * H3;
      for (int rr = tid; rr < H3; rr += NT) __builtin_prefetch(nx + rr, 0, 1);
    }
#endif
    __syncthreads();
#if HAVE_TDM
    if (((tid >> 5) == 0) && (t + 1 < TT))                 // async load t+1
      tdm_load_col(ig0pre + (size_t)(t + 1) * H3,
                   ig_lds0 + (unsigned)(((t + 1) & 1) * H3 * 4));
    const float* igc = igbuf + (size_t)(t & 1) * H3;
#else
    const float* igc = ig0pre + (size_t)t * H3;
#endif

    // -------- GRU0: ig (precomputed + state cols + bias) and hg mat-vecs ----
    for (int r0 = 0; r0 < 5; ++r0) {
      int rr = tid + r0 * NT;
      if (rr < H3) {
        float aI = igc[rr] + b0[rr];
        for (int k = 0; k < 152; ++k)
          aI += bf_bits2f(w0ihT[(size_t)(NRAW + k) * H3 + rr]) * inps[k];
        float aH = 0.f;
        for (int k = 0; k < HID; ++k)
          aH += bf_bits2f(w0hhT[(size_t)k * H3 + rr]) * h0[k];
        bufA[rr] = aI; bufB[rr] = aH;
      }
    }
    __syncthreads();
    for (int j = tid; j < HID; j += NT) {
      float r = 1.f / (1.f + __expf(-(bufA[j] + bufB[j])));
      float z = 1.f / (1.f + __expf(-(bufA[j + HID] + bufB[j + HID])));
      float n = tanhf(bufA[j + 2 * HID] + r * (bufB[j + 2 * HID] + bn0[j]));
      h0[j] = n + z * (h0[j] - n);
    }
    __syncthreads();

    // -------- GRU1 ----------------------------------------------------------
    for (int r0 = 0; r0 < 5; ++r0) {
      int rr = tid + r0 * NT;
      if (rr < H3) {
        float aI = b1[rr], aH = 0.f;
        for (int k = 0; k < HID; ++k) {
          aI += bf_bits2f(w1ihT[(size_t)k * H3 + rr]) * h0[k];
          aH += bf_bits2f(w1hhT[(size_t)k * H3 + rr]) * h1[k];
        }
        bufA[rr] = aI; bufB[rr] = aH;
      }
    }
    __syncthreads();
    for (int j = tid; j < HID; j += NT) {
      float r = 1.f / (1.f + __expf(-(bufA[j] + bufB[j])));
      float z = 1.f / (1.f + __expf(-(bufA[j + HID] + bufB[j + HID])));
      float n = tanhf(bufA[j + 2 * HID] + r * (bufB[j + 2 * HID] + bn1[j]));
      h1[j] = n + z * (h1[j] - n);
    }
    __syncthreads();

    // -------- linear head + theta_feat concat -------------------------------
    for (int o = tid; o < OUTD; o += NT) {
      float acc = lin_b[o];
      for (int k = 0; k < HID; ++k)
        acc += bf_bits2f(linwT[(size_t)k * OUTD + o]) * h1[k];
      outv[o] = acc;
      outv[OUTD + o] = tf[o];
    }
    __syncthreads();

    // -------- layer norm over 6528 ------------------------------------------
    float s = 0.f;
    for (int j = tid; j < LN_N; j += NT) s += outv[j];
    red[tid] = s; __syncthreads();
    for (int o = NT / 2; o > 0; o >>= 1) { if (tid < o) red[tid] += red[tid + o]; __syncthreads(); }
    float mu = red[0] / (float)LN_N;
    __syncthreads();
    float s2 = 0.f;
    for (int j = tid; j < LN_N; j += NT) { float d = outv[j] - mu; s2 += d * d; }
    red[tid] = s2; __syncthreads();
    for (int o = NT / 2; o > 0; o >>= 1) { if (tid < o) red[tid] += red[tid + o]; __syncthreads(); }
    float inv = rsqrtf(red[0] / (float)LN_N + 1e-5f);
    __syncthreads();
    for (int j = tid; j < LN_N; j += NT)
      outv[j] = (outv[j] - mu) * inv * flnw[j] + flnb[j];
    __syncthreads();

    // -------- f1 relu -------------------------------------------------------
    for (int h = tid; h < HID; h += NT) {
      float acc = f1b[h];
      for (int k = 0; k < LN_N; ++k)
        acc += bf_bits2f(f1wT[(size_t)k * HID + h]) * outv[k];
      hidv[h] = fmaxf(acc, 0.f);
    }
    __syncthreads();

    // -------- f2 (only first P2=1632 outputs are consumed) ------------------
    for (int o = tid; o < HID; o += NT) {
      float acc = f2b[o];
      for (int k = 0; k < HID; ++k)
        acc += bf_bits2f(f2wT[(size_t)k * OUTD + o]) * hidv[k];
      out2[o] = acc;
    }
    __syncthreads();

    // -------- Kalman residual blocks (fp32) ---------------------------------
    for (int r = 0; r < NRES; ++r) {
      const float* kw = out2 + 64 + r * 256;          // wgt_state[r]
      if (tid < 256) {                                // L (lower) from chol params
        int i = tid >> 4, j = tid & 15;
        float v = 0.f;
        if (j <= i) {
          int a = j, b = i;                           // U[a][b], a<=b
          int uidx = 16 * a - (a * (a - 1)) / 2 + (b - a);
          float cc = out2[1088 + r * NTRI_ + uidx];
          v = (j == i) ? ((cc > 20.f) ? cc : log1pf(__expf(cc))) : cc;
        }
        Lr_[tid] = v;
      }
      __syncthreads();
      if (tid < 256) {                                // var_state = L L^T
        int i = tid >> 4, j = tid & 15;
        float acc = 0.f;
        for (int k = 0; k < 16; ++k) acc += Lr_[i * 16 + k] * Lr_[j * 16 + k];
        vs_[tid] = acc;
      }
      if (tid < 16) {                                 // mp = kw @ m + bias (out2[r*16+i])
        float acc = out2[r * 16 + tid];
        for (int k = 0; k < 16; ++k) acc += kw[tid * 16 + k] * km_[k];
        mp_[tid] = acc;
      }
      __syncthreads();
      if (tid < 256) {                                // t1 = kw @ v
        int i = tid >> 4, j = tid & 15;
        float acc = 0.f;
        for (int k = 0; k < 16; ++k) acc += kw[i * 16 + k] * kv_[k * 16 + j];
        t1_[tid] = acc;
      }
      __syncthreads();
      if (tid < 256) {                                // vp = t1 @ kw^T + vs
        int i = tid >> 4, j = tid & 15;
        float acc = vs_[tid];
        for (int k = 0; k < 16; ++k) acc += t1_[i * 16 + k] * kw[j * 16 + k];
        vp_[tid] = acc;
      }
      __syncthreads();
      size_t row = (size_t)4 * t + r;
      if (tid < 256) wg[row * 256 + tid] = kw[tid];
      if (r < NRES - 1) {
        if (tid < 16)  { pm[row * 16 + tid] = mp_[tid]; fm[(row + 1) * 16 + tid] = mp_[tid]; km_[tid] = mp_[tid]; }
        if (tid < 256) { pv[row * 256 + tid] = vp_[tid]; fv[(row + 1) * 256 + tid] = vp_[tid]; kv_[tid] = vp_[tid]; }
        __syncthreads();
      } else {
        // measurement update
        if (tid < 128) {                              // C = vp @ H^T
          int i = tid >> 3, a = tid & 7;
          float acc = 0.f;
          for (int k = 0; k < 16; ++k) acc += vp_[i * 16 + k] * Hm[a * 16 + k];
          C_[tid] = acc;
        }
        if (tid < 8) {                                // resid = y - H mp
          float acc = ym[(size_t)t * NMEAS + tid];
          for (int k = 0; k < 16; ++k) acc -= Hm[tid * 16 + k] * mp_[k];
          rs_[tid] = acc;
        }
        __syncthreads();
        if (tid < 64) {                               // S = H @ C + R
          int a = tid >> 3, b = tid & 7;
          float acc = Rm[tid];
          for (int k = 0; k < 16; ++k) acc += Hm[a * 16 + k] * C_[k * 8 + b];
          S_[tid] = acc;
        }
        __syncthreads();
        if (tid == 0) {                               // K = C S^{-1} (SPD, GJ)
          float A[8][8], X[8][16];
          for (int a = 0; a < 8; ++a) {
            for (int b = 0; b < 8; ++b) A[a][b] = S_[a * 8 + b];
            for (int i = 0; i < 16; ++i) X[a][i] = C_[i * 8 + a];
          }
          for (int p = 0; p < 8; ++p) {
            float ip = 1.f / A[p][p];
            for (int b = p; b < 8; ++b) A[p][b] *= ip;
            for (int i = 0; i < 16; ++i) X[p][i] *= ip;
            for (int a = 0; a < 8; ++a) {
              if (a == p) continue;
              float f = A[a][p];
              for (int b = p; b < 8; ++b) A[a][b] -= f * A[p][b];
              for (int i = 0; i < 16; ++i) X[a][i] -= f * X[p][i];
            }
          }
          for (int i = 0; i < 16; ++i)
            for (int a = 0; a < 8; ++a) Kg_[i * 8 + a] = X[a][i];
        }
        __syncthreads();
        if (tid < 16) {                               // m_next = mp + K resid
          float acc = mp_[tid];
          for (int a = 0; a < 8; ++a) acc += Kg_[tid * 8 + a] * rs_[a];
          mn_[tid] = acc;
        }
        if (tid < 256) {                              // v_next = vp - K C^T
          int i = tid >> 4, j = tid & 15;
          float acc = vp_[tid];
          for (int a = 0; a < 8; ++a) acc -= Kg_[i * 8 + a] * C_[j * 8 + a];
          vn_[tid] = acc;
        }
        __syncthreads();
        if (tid < 16)  { pm[row * 16 + tid] = mp_[tid]; fm[(row + 1) * 16 + tid] = mn_[tid]; km_[tid] = mn_[tid]; }
        if (tid < 256) { pv[row * 256 + tid] = vp_[tid]; fv[(row + 1) * 256 + tid] = vn_[tid]; kv_[tid] = vn_[tid]; }
        __syncthreads();
      }
    }
  }
}

// ---------------------------------------------------------------------------
extern "C" void kernel_launch(void* const* d_in, const int* in_sizes, int n_in,
                              void* d_out, int out_size, void* d_ws, size_t ws_size,
                              hipStream_t stream) {
  (void)in_sizes; (void)n_in; (void)out_size; (void)ws_size;
  const float* data_seq = (const float*)d_in[0];
  const float* theta    = (const float*)d_in[1];
  const float* msi      = (const float*)d_in[2];
  const float* vsi      = (const float*)d_in[3];
  const float* wgt_meas = (const float*)d_in[4];
  const float* var_meas = (const float*)d_in[5];
  const float* y_meas   = (const float*)d_in[6];
  const float* g0_wih   = (const float*)d_in[7];
  const float* g0_whh   = (const float*)d_in[8];
  const float* g0_b     = (const float*)d_in[9];
  const float* g0_bn    = (const float*)d_in[10];
  const float* g1_wih   = (const float*)d_in[11];
  const float* g1_whh   = (const float*)d_in[12];
  const float* g1_b     = (const float*)d_in[13];
  const float* g1_bn    = (const float*)d_in[14];
  const float* l0_w     = (const float*)d_in[15];
  const float* l0_b     = (const float*)d_in[16];
  const float* l1_w     = (const float*)d_in[17];
  const float* l1_b     = (const float*)d_in[18];
  const float* lin_w    = (const float*)d_in[19];
  const float* lin_b    = (const float*)d_in[20];
  const float* fln_w    = (const float*)d_in[21];
  const float* fln_b    = (const float*)d_in[22];
  const float* f1_w     = (const float*)d_in[23];
  const float* f1_b     = (const float*)d_in[24];
  const float* f2_w     = (const float*)d_in[25];
  const float* f2_b     = (const float*)d_in[26];
  float* out = (float*)d_out;

  // ---- workspace carve-out ----
  char* ws = (char*)d_ws;
  size_t off = 0;
  auto carve = [&](size_t bytes) -> void* {
    void* p = ws + off; off += (bytes + 255) & ~(size_t)255; return p;
  };
  unsigned short* w0ihT = (unsigned short*)carve((size_t)256 * H3 * 2);
  unsigned short* w0hhT = (unsigned short*)carve((size_t)HID * H3 * 2);
  unsigned short* w1ihT = (unsigned short*)carve((size_t)HID * H3 * 2);
  unsigned short* w1hhT = (unsigned short*)carve((size_t)HID * H3 * 2);
  unsigned short* linwT = (unsigned short*)carve((size_t)HID * OUTD * 2);
  unsigned short* f1wT  = (unsigned short*)carve((size_t)LN_N * HID * 2);
  unsigned short* f2wT  = (unsigned short*)carve((size_t)HID * OUTD * 2);
  unsigned short* aG    = (unsigned short*)carve((size_t)H3 * KPAD * 2);  // GEMM A staged
  unsigned short* bG    = (unsigned short*)carve((size_t)TT * KPAD * 2);  // GEMM B staged
  float* ig0pre = (float*)carve((size_t)TT * H3 * 4);
  float* tf     = (float*)carve((size_t)OUTD * 4);

  auto grid_of = [](size_t n) { return (unsigned)((n + 255) / 256); };

  // ---- one-time weight staging to bf16 (transposed for the scan) ----
  conv_tr_kernel<<<grid_of((size_t)H3 * 256), 256, 0, stream>>>(g0_wih, w0ihT, H3, 256);
  conv_tr_kernel<<<grid_of((size_t)H3 * HID), 256, 0, stream>>>(g0_whh, w0hhT, H3, HID);
  conv_tr_kernel<<<grid_of((size_t)H3 * HID), 256, 0, stream>>>(g1_wih, w1ihT, H3, HID);
  conv_tr_kernel<<<grid_of((size_t)H3 * HID), 256, 0, stream>>>(g1_whh, w1hhT, H3, HID);
  conv_tr_kernel<<<grid_of((size_t)OUTD * HID), 256, 0, stream>>>(lin_w, linwT, OUTD, HID);
  conv_tr_kernel<<<grid_of((size_t)HID * LN_N), 256, 0, stream>>>(f1_w, f1wT, HID, LN_N);
  conv_tr_kernel<<<grid_of((size_t)OUTD * HID), 256, 0, stream>>>(f2_w, f2wT, OUTD, HID);

  // ---- K-padded row-major bf16 GEMM operands ----
  stage_pad_kernel<<<grid_of((size_t)H3 * KPAD), 256, 0, stream>>>(g0_wih, aG, H3, 256, KPAD, NRAW);
  stage_pad_kernel<<<grid_of((size_t)TT * KPAD), 256, 0, stream>>>(data_seq, bG, TT, NRAW, KPAD, NRAW);

  // ---- theta features (one-time) ----
  theta_feat_kernel<<<1, 1024, 0, stream>>>(theta, l0_w, l0_b, l1_w, l1_b, tf);

  // ---- WMMA GEMM: time-batched input-gate precompute ----
  {
    unsigned waves = GEMM_MT * GEMM_NT;          // 38250 16x16 tiles
    unsigned blocks = (waves + 7) / 8;           // 8 waves / 256-thread block
    ig0_gemm_kernel<<<blocks, 256, 0, stream>>>(aG, bG, ig0pre);
  }

  // ---- persistent sequential scan ----
  const size_t smem_floats =
      2 * HID + 2 * H3 + LN_N + HID + HID + 160 + NT +
      (16 + 256 + 16 + 256 + 256 + 256 + 256 + 64 + 128 + 128 + 8 + 16 + 256) +
      2 * H3;                                            // TDM double buffer
  const size_t smem_bytes = smem_floats * sizeof(float); // ~143 KB (WGP: 320KB)
  (void)hipFuncSetAttribute((const void*)rnn_scan_kernel,
                            hipFuncAttributeMaxDynamicSharedMemorySize,
                            (int)smem_bytes);
  rnn_scan_kernel<<<1, NT, smem_bytes, stream>>>(
      ig0pre, w0ihT, w0hhT, w1ihT, w1hhT, linwT, f1wT, f2wT, tf,
      g0_b, g0_bn, g1_b, g1_bn, lin_b, fln_w, fln_b, f1_b, f2_b,
      wgt_meas, var_meas, y_meas, msi, vsi, out);
}
